// Belt_Block_83665962926555
// MI455X (gfx1250) — matrix operand order repaired and checked by reading, not compiled, not run on hardware
//
#include <hip/hip_runtime.h>
#include <hip/hip_bf16.h>

typedef __attribute__((ext_vector_type(16))) _Float16 v16h;
typedef __attribute__((ext_vector_type(8)))  _Float16 v8h;
typedef __attribute__((ext_vector_type(8)))  float    v8f;

namespace {
constexpr int Bn = 8, Nn = 307, FINn = 64, Tn = 12;
constexpr int HSn = 4, COn = 16, Cn = 64, HTn = 4, DKn = 16, FFn = 256;
constexpr float EPSn = 1e-5f;
constexpr int BT = Bn * Tn;                 // 96
constexpr int TILES_BT = (Nn + 15) / 16;    // 20 row-tiles per (b,t)
constexpr int NP = 320;                     // padded rows per (b,t) plane
constexpr int NS = Nn * Cn * Tn;            // 235776 valid elements / batch
constexpr int TNPC = Tn * NP * Cn;          // padded plane stride per batch
constexpr size_t PSZ = (size_t)BT * NP * Cn;  // 1966080 floats per buffer
constexpr size_t SZQ = (size_t)Bn * Nn * HTn * Tn * DKn;  // 1886208
constexpr int MFF = Bn * Nn * Tn;           // 29472 FFN rows
constexpr int MFT = (MFF + 15) / 16;        // 1842 tiles
}

__device__ inline v8f wmma_f16(v16h a, v16h b, v8f c) {
  return __builtin_amdgcn_wmma_f32_16x16x32_f16(false, a, false, b, (short)0, c,
                                                false, false);
}

// A fragment from contiguous f16 row (two 16B loads per lane).
__device__ inline v16h load_a_f16row(const _Float16* rowp, int kk, int lane) {
  int kgrp = (lane >> 4) << 3;
  const v8h* p = (const v8h*)(rowp + kk + kgrp);
  v8h lo = p[0];
  v8h hi = p[2];  // +16 halves
  v16h a;
#pragma unroll
  for (int h = 0; h < 8; ++h) { a[h] = lo[h]; a[8 + h] = hi[h]; }
  return a;
}

// A fragment from contiguous f32 row (cvt on the fly).
__device__ inline v16h load_a_f32row(const float* rowp, int kk, int lane) {
  int kgrp = (lane >> 4) << 3;
  const float* p0 = rowp + kk + kgrp;
  v16h a;
#pragma unroll
  for (int h = 0; h < 8; ++h) {
    a[h] = (_Float16)p0[h];
    a[8 + h] = (_Float16)p0[16 + h];
  }
  return a;
}

// Pre-swizzled packed B fragment: frag (ct,ks), lane-major, 16 halves each.
__device__ inline v16h load_b_pack(const _Float16* base, int ksteps, int ct,
                                   int ks, int lane) {
  return ((const v16h*)base)[(ct * ksteps + ks) * 32 + lane];
}

// ---------------- weight packing: f32 (Cout x K) -> f16 WMMA B fragments ----
__global__ __launch_bounds__(256) void k_packw(const float* __restrict__ W,
                                               int ldw, int tiles, int ksteps,
                                               _Float16* __restrict__ dst) {
  int tid = blockIdx.x * blockDim.x + threadIdx.x;
  int total = tiles * ksteps * 32;
  if (tid >= total) return;
  int lane = tid & 31;
  int f = tid >> 5;
  int ks = f % ksteps, ct = f / ksteps;
  // B[k,c] = W[c*ldw + k]; lane col = ct*16+(lane&15); h -> k = ks*32+(lane>>4)*8+h (+8 for h>=8)
  const float* wrow =
      W + (size_t)(ct * 16 + (lane & 15)) * ldw + ks * 32 + ((lane >> 4) << 3);
  _Float16* o = dst + (size_t)tid * 16;
#pragma unroll
  for (int h = 0; h < 16; ++h) o[h] = (_Float16)wrow[h + ((h >> 3) << 3)];
}

// ---------------- pack x: (B,N,FIN,T) f32 -> padded (BT,NP,FIN) f16 ---------
__global__ __launch_bounds__(256) void k_pack_x(const float* __restrict__ x,
                                                _Float16* __restrict__ xt) {
  size_t tid = (size_t)blockIdx.x * 256 + threadIdx.x;
  if (tid >= (size_t)BT * NP * FINn) return;
  int f = (int)(tid % FINn);
  int n = (int)((tid / FINn) % NP);
  int bt = (int)(tid / ((size_t)FINn * NP));
  int b = bt / Tn, t = bt % Tn;
  float v = (n < Nn) ? x[((size_t)(b * Nn + n) * FINn + f) * Tn + t] : 0.f;
  xt[tid] = (_Float16)v;
}

// ---------------- prep: sparse neighbor lists -------------------------------
__global__ __launch_bounds__(128) void k_prep(const float* __restrict__ adj,
                                              const float* __restrict__ iw,
                                              int* __restrict__ nbri,
                                              float* __restrict__ nbrw) {
  int i = blockIdx.x * blockDim.x + threadIdx.x;
  if (i >= Nn) return;
  int jj[3] = {i, i, i};
  float wv[3] = {0.f, 0.f, 0.f};
  int cnt = 0;
  float s = 0.f;
  for (int j = 0; j < Nn && cnt < 3; ++j) {
    if (adj[(size_t)i * Nn + j] > 0.5f) {
      float w = iw[(size_t)i * Nn + j];
      w = w > 0.f ? w : 0.f;
      jj[cnt] = j; wv[cnt] = w; s += w; ++cnt;
    }
  }
  if (s == 0.f) s = 1e-6f;
  float invs = 1.f / s;
#pragma unroll
  for (int k = 0; k < 3; ++k) {
    nbri[i * 3 + k] = jj[k];
    nbrw[i * 3 + k] = wv[k] * invs;
  }
}

// ---------------- GEMM: Wh = xp @ W_w^T + b  (f16 A, packed B) --------------
__global__ __launch_bounds__(128) void k_gemm_wh(
    const _Float16* __restrict__ xt, const _Float16* __restrict__ pW,
    const float* __restrict__ Wb, float* __restrict__ Wh) {
  int wave = threadIdx.x >> 5, lane = threadIdx.x & 31;
  if (threadIdx.x == 0) __builtin_prefetch(pW, 0, 3);
  int g = blockIdx.x * 4 + wave;  // exactly BT*20 tiles
  int bt = g / TILES_BT, tb = g % TILES_BT;
  int n0 = tb * 16;
  const _Float16* rowp = xt + ((size_t)bt * NP + n0 + (lane & 15)) * FINn;
  v16h a0 = load_a_f16row(rowp, 0, lane);
  v16h a1 = load_a_f16row(rowp, 32, lane);
  int ccol = lane & 15;
#pragma unroll
  for (int ct = 0; ct < 4; ++ct) {
    float bias = Wb[ct * 16 + ccol];
    v8f acc;
#pragma unroll
    for (int r = 0; r < 8; ++r) acc[r] = bias;
    acc = wmma_f16(a0, load_b_pack(pW, 2, ct, 0, lane), acc);
    acc = wmma_f16(a1, load_b_pack(pW, 2, ct, 1, lane), acc);
    // Wh is padded: store unconditionally (pad rows are scratch)
#pragma unroll
    for (int r = 0; r < 8; ++r) {
      int nn = n0 + r + ((lane >> 4) << 3);
      Wh[((size_t)bt * NP + nn) * Cn + ct * 16 + ccol] = acc[r];
    }
  }
}

// ---------------- column sums S[b,t,c] = sum_n Wh ---------------------------
__global__ __launch_bounds__(64) void k_redS(const float* __restrict__ Wh,
                                             float* __restrict__ S) {
  int bt = blockIdx.x, c = threadIdx.x;
  float s = 0.f;
  for (int n = 0; n < Nn; ++n) s += Wh[((size_t)bt * NP + n) * Cn + c];
  S[bt * Cn + c] = s;
}

// ---------------- s1/s2 head scores -----------------------------------------
__global__ __launch_bounds__(256) void k_scores(const float* __restrict__ Wh,
                                                const float* __restrict__ aa,
                                                float* __restrict__ s1,
                                                float* __restrict__ s2) {
  int tid = blockIdx.x * blockDim.x + threadIdx.x;
  if (tid >= BT * Nn * HSn) return;
  int h = tid % HSn;
  int n = (tid / HSn) % Nn;
  int bt = tid / (HSn * Nn);
  const float* whp = Wh + ((size_t)bt * NP + n) * Cn + h * COn;
  float a = 0.f, d = 0.f;
#pragma unroll
  for (int c = 0; c < COn; ++c) {
    float v = whp[c];
    a += v * aa[h * 32 + c];
    d += v * aa[h * 32 + 16 + c];
  }
  s1[tid] = a;
  s2[tid] = d;
}

// ---------------- sparse GAT aggregation + residual -> y1pre (bt,n,c) -------
__global__ __launch_bounds__(256) void k_gat(
    const float* __restrict__ Wh, const float* __restrict__ S,
    const float* __restrict__ s1, const float* __restrict__ s2,
    const int* __restrict__ nbri, const float* __restrict__ nbrw,
    const float* __restrict__ x, float* __restrict__ y1pre) {
  int tid = blockIdx.x * blockDim.x + threadIdx.x;
  if (tid >= BT * Nn * HSn) return;
  int h = tid % HSn;
  int i = (tid / HSn) % Nn;
  int bt = tid / (HSn * Nn);
  int b = bt / Tn, t = bt % Tn;
  float s1i = s1[tid];
  int jidx[3]; float sc[3];
#pragma unroll
  for (int k = 0; k < 3; ++k) {
    int j = nbri[i * 3 + k];
    jidx[k] = j;
    float z = s1i + s2[((size_t)bt * Nn + j) * HSn + h];
    float lr = z > 0.f ? z : 0.2f * z;  // leaky_relu(0.2)
    sc[k] = lr + nbrw[i * 3 + k];
  }
  float m = 0.f;
#pragma unroll
  for (int k = 0; k < 3; ++k) m = fmaxf(m, sc[k]);
  float e0 = __expf(-m);
  float Z = (float)(Nn - 3) * e0;
  float ek[3];
#pragma unroll
  for (int k = 0; k < 3; ++k) { ek[k] = __expf(sc[k] - m); Z += ek[k]; }
  float inv = 1.f / Z;
  const float* Sp = S + bt * Cn + h * COn;
  const float* whb = Wh + (size_t)bt * NP * Cn;
#pragma unroll
  for (int d = 0; d < COn; ++d) {
    float acc = e0 * Sp[d];
#pragma unroll
    for (int k = 0; k < 3; ++k)
      acc += (ek[k] - e0) * whb[(size_t)jidx[k] * Cn + h * COn + d];
    y1pre[((size_t)bt * NP + i) * Cn + h * COn + d] =
        acc * inv + x[((size_t)(b * Nn + i) * Cn + h * COn + d) * Tn + t];
  }
}

// ---------------- LayerNorm stats (padded src), two-stage deterministic -----
__global__ __launch_bounds__(256) void k_stats(const float* __restrict__ src,
                                               float* __restrict__ part) {
  int b = blockIdx.y;
  const float* p = src + (size_t)b * TNPC;
  float s = 0.f, q = 0.f;
  for (int i = blockIdx.x * 256 + threadIdx.x; i < NS; i += gridDim.x * 256) {
    int t = i / (Nn * Cn);
    int rem = i - t * (Nn * Cn);
    int n = rem / Cn;
    int c = rem - n * Cn;
    float v = p[((size_t)t * NP + n) * Cn + c];
    s += v; q += v * v;
  }
  __shared__ float r1[256], r2[256];
  r1[threadIdx.x] = s; r2[threadIdx.x] = q;
  __syncthreads();
  for (int o = 128; o > 0; o >>= 1) {
    if (threadIdx.x < o) {
      r1[threadIdx.x] += r1[threadIdx.x + o];
      r2[threadIdx.x] += r2[threadIdx.x + o];
    }
    __syncthreads();
  }
  if (threadIdx.x == 0) {
    part[(b * 64 + blockIdx.x) * 2 + 0] = r1[0];
    part[(b * 64 + blockIdx.x) * 2 + 1] = r2[0];
  }
}

__global__ __launch_bounds__(64) void k_stats_fin(const float* __restrict__ part,
                                                  float* __restrict__ stats) {
  int b = blockIdx.x;
  __shared__ float r1[64], r2[64];
  r1[threadIdx.x] = part[(b * 64 + threadIdx.x) * 2 + 0];
  r2[threadIdx.x] = part[(b * 64 + threadIdx.x) * 2 + 1];
  __syncthreads();
  for (int o = 32; o > 0; o >>= 1) {
    if (threadIdx.x < o) {
      r1[threadIdx.x] += r1[threadIdx.x + o];
      r2[threadIdx.x] += r2[threadIdx.x + o];
    }
    __syncthreads();
  }
  if (threadIdx.x == 0) { stats[2 * b] = r1[0]; stats[2 * b + 1] = r2[0]; }
}

// mode 0: write padded (bt,n,c); mode 1: write d_out in (B,N,C,T)
__global__ __launch_bounds__(256) void k_apply_ln(
    const float* __restrict__ src, const float* __restrict__ g,
    const float* __restrict__ be, const float* __restrict__ stats,
    float* __restrict__ dst, int mode) {
  size_t tid = (size_t)blockIdx.x * 256 + threadIdx.x;
  if (tid >= (size_t)Bn * NS) return;
  int b = (int)(tid / NS);
  int i = (int)(tid % NS);
  int t = i / (Nn * Cn);
  int rem = i - t * (Nn * Cn);
  int n = rem / Cn;
  int c = rem - n * Cn;
  float invN = 1.f / (float)NS;
  float mean = stats[2 * b] * invN;
  float var = stats[2 * b + 1] * invN - mean * mean;
  size_t a = (size_t)b * TNPC + ((size_t)t * NP + n) * Cn + c;
  int gi = (n * Cn + c) * Tn + t;  // g/be are (N,C,T)
  float v = (src[a] - mean) * rsqrtf(var + EPSn) * g[gi] + be[gi];
  if (mode == 0)
    dst[a] = v;
  else
    dst[((size_t)(b * Nn + n) * Cn + c) * Tn + t] = v;
}

// ---------------- Q/K/V projections (packed B, contiguous f32 A) ------------
__global__ __launch_bounds__(128) void k_qkv(
    const float* __restrict__ y1, const _Float16* __restrict__ pWq,
    const _Float16* __restrict__ pWk, const _Float16* __restrict__ pWv,
    const float* __restrict__ Wqb, const float* __restrict__ Wkb,
    const float* __restrict__ Wvb, float* __restrict__ Q,
    float* __restrict__ Kb, float* __restrict__ Vb) {
  const _Float16* pW; const float* bias; float* dst;
  if (blockIdx.y == 0)      { pW = pWq; bias = Wqb; dst = Q; }
  else if (blockIdx.y == 1) { pW = pWk; bias = Wkb; dst = Kb; }
  else                      { pW = pWv; bias = Wvb; dst = Vb; }
  int wave = threadIdx.x >> 5, lane = threadIdx.x & 31;
  if (threadIdx.x == 0) __builtin_prefetch(pW, 0, 3);
  int g = blockIdx.x * 4 + wave;
  int bt = g / TILES_BT, tb = g % TILES_BT;
  int b = bt / Tn, t = bt % Tn;
  int n0 = tb * 16;
  const float* rowp = y1 + ((size_t)bt * NP + n0 + (lane & 15)) * Cn;
  v16h a0 = load_a_f32row(rowp, 0, lane);
  v16h a1 = load_a_f32row(rowp, 32, lane);
  int ccol = lane & 15;
  bool full = (n0 + 16 <= Nn);
#pragma unroll
  for (int ct = 0; ct < 4; ++ct) {  // ct == head (DK==16)
    float bv = bias[ct * 16 + ccol];
    v8f acc;
#pragma unroll
    for (int r = 0; r < 8; ++r) acc[r] = bv;
    acc = wmma_f16(a0, load_b_pack(pW, 2, ct, 0, lane), acc);
    acc = wmma_f16(a1, load_b_pack(pW, 2, ct, 1, lane), acc);
    if (full) {
#pragma unroll
      for (int r = 0; r < 8; ++r) {
        int nn = n0 + r + ((lane >> 4) << 3);
        dst[(((size_t)(b * Nn + nn) * HTn + ct) * Tn + t) * DKn + ccol] =
            acc[r];
      }
    } else {
#pragma unroll
      for (int r = 0; r < 8; ++r) {
        int nn = n0 + r + ((lane >> 4) << 3);
        if (nn < Nn)
          dst[(((size_t)(b * Nn + nn) * HTn + ct) * Tn + t) * DKn + ccol] =
              acc[r];
      }
    }
  }
}

// ---------------- tiny temporal attention (T=12, DK=16) ---------------------
__global__ __launch_bounds__(256) void k_attn(const float* __restrict__ Q,
                                              const float* __restrict__ Kb,
                                              const float* __restrict__ Vb,
                                              float* __restrict__ ctx) {
  int tid = blockIdx.x * blockDim.x + threadIdx.x;
  if (tid >= Bn * Nn * HTn * Tn) return;
  int t = tid % Tn;
  int h = (tid / Tn) % HTn;
  int n = (tid / (Tn * HTn)) % Nn;
  int b = tid / (Tn * HTn * Nn);
  size_t base = (((size_t)(b * Nn + n) * HTn + h) * Tn) * DKn;
  const float* qp = Q + base + (size_t)t * DKn;
  const float* kp = Kb + base;
  const float* vp = Vb + base;
  float q[DKn];
#pragma unroll
  for (int d = 0; d < DKn; ++d) q[d] = qp[d];
  float sc[Tn];
  float mx = -1e30f;
#pragma unroll
  for (int s = 0; s < Tn; ++s) {
    float d = 0.f;
#pragma unroll
    for (int dd = 0; dd < DKn; ++dd) d += q[dd] * kp[s * DKn + dd];
    d *= 0.25f;  // 1/sqrt(DK)
    sc[s] = d;
    mx = fmaxf(mx, d);
  }
  float Z = 0.f;
#pragma unroll
  for (int s = 0; s < Tn; ++s) { sc[s] = __expf(sc[s] - mx); Z += sc[s]; }
  float inv = 1.f / Z;
  float o[DKn];
#pragma unroll
  for (int d = 0; d < DKn; ++d) o[d] = 0.f;
#pragma unroll
  for (int s = 0; s < Tn; ++s) {
    float a = sc[s] * inv;
#pragma unroll
    for (int d = 0; d < DKn; ++d) o[d] += a * vp[s * DKn + d];
  }
  float* op = ctx + ((size_t)(b * Tn + t) * NP + n) * Cn + h * DKn;
#pragma unroll
  for (int d = 0; d < DKn; ++d) op[d] = o[d];
}

// ---------------- O projection + residual -> y2pre (padded) -----------------
__global__ __launch_bounds__(128) void k_gemm_o(
    const float* __restrict__ ctx, const _Float16* __restrict__ pW,
    const float* __restrict__ Wob, const float* __restrict__ y1,
    float* __restrict__ y2pre) {
  int wave = threadIdx.x >> 5, lane = threadIdx.x & 31;
  if (threadIdx.x == 0) __builtin_prefetch(pW, 0, 3);
  int g = blockIdx.x * 4 + wave;
  int bt = g / TILES_BT, tb = g % TILES_BT;
  int n0 = tb * 16;
  const float* rowp = ctx + ((size_t)bt * NP + n0 + (lane & 15)) * Cn;
  v16h a0 = load_a_f32row(rowp, 0, lane);
  v16h a1 = load_a_f32row(rowp, 32, lane);
  int ccol = lane & 15;
#pragma unroll
  for (int ct = 0; ct < 4; ++ct) {
    float bv = Wob[ct * 16 + ccol];
    v8f acc;
#pragma unroll
    for (int r = 0; r < 8; ++r) acc[r] = bv;
    acc = wmma_f16(a0, load_b_pack(pW, 2, ct, 0, lane), acc);
    acc = wmma_f16(a1, load_b_pack(pW, 2, ct, 1, lane), acc);
    // padded dst: unconditional stores; pad rows are scratch
#pragma unroll
    for (int r = 0; r < 8; ++r) {
      int nn = n0 + r + ((lane >> 4) << 3);
      size_t o = ((size_t)bt * NP + nn) * Cn + ct * 16 + ccol;
      y2pre[o] = acc[r] + y1[o];
    }
  }
}

// ---------------- fused FFN with async-LDS staged packed weights ------------
__global__ __launch_bounds__(128) void k_ff(const float* __restrict__ y2,
                                            const _Float16* __restrict__ pW1,
                                            const float* __restrict__ b1,
                                            const _Float16* __restrict__ pW2,
                                            const float* __restrict__ b2,
                                            float* __restrict__ y3pre) {
  __shared__ __align__(32) _Float16 W1s[16384];  // 32 KB packed fragments
  __shared__ __align__(32) _Float16 W2s[16384];  // 32 KB packed fragments
  __shared__ _Float16 Hb[4][16][FFn + 8];
  int wave = threadIdx.x >> 5, lane = threadIdx.x & 31;

  // Stage both FFN weight fragment arrays into LDS with the CDNA5 async path.
  {
    unsigned l1 = (unsigned)(uintptr_t)(&W1s[0]);
    unsigned l2 = (unsigned)(uintptr_t)(&W2s[0]);
    unsigned long long g1 = (unsigned long long)(uintptr_t)pW1;
    unsigned long long g2 = (unsigned long long)(uintptr_t)pW2;
#pragma unroll 1
    for (int it = 0; it < 16; ++it) {
      unsigned chunk = (unsigned)(it * 128 + threadIdx.x) * 16u;  // 16B chunks
      asm volatile("global_load_async_to_lds_b128 %0, %1, off"
                   :: "v"(l1 + chunk), "v"(g1 + chunk) : "memory");
      asm volatile("global_load_async_to_lds_b128 %0, %1, off"
                   :: "v"(l2 + chunk), "v"(g2 + chunk) : "memory");
    }
    asm volatile("s_wait_asynccnt 0" ::: "memory");
    __syncthreads();
  }

  int g = blockIdx.x * 4 + wave;
  int m_base = g * 16;
  int m = m_base + (lane & 15);
  bool valid = (m < MFF);
  int mm = valid ? m : 0;
  int b = mm / (Nn * Tn);
  int rem = mm % (Nn * Tn);
  int n = rem / Tn, t = rem % Tn;
  const float* rowp = y2 + ((size_t)(b * Tn + t) * NP + n) * Cn;  // contiguous
  v16h a0 = load_a_f32row(rowp, 0, lane);
  v16h a1 = load_a_f32row(rowp, 32, lane);
  int ccol = lane & 15;
  _Float16(*H)[FFn + 8] = Hb[wave];
  // GEMM1: (16x64) @ (64x256), ReLU -> LDS hidden
#pragma unroll
  for (int ct = 0; ct < 16; ++ct) {
    float bv = b1[ct * 16 + ccol];
    v8f acc;
#pragma unroll
    for (int r = 0; r < 8; ++r) acc[r] = bv;
    acc = wmma_f16(a0, load_b_pack(W1s, 2, ct, 0, lane), acc);
    acc = wmma_f16(a1, load_b_pack(W1s, 2, ct, 1, lane), acc);
#pragma unroll
    for (int r = 0; r < 8; ++r) {
      int row = r + ((lane >> 4) << 3);
      float v = acc[r] > 0.f ? acc[r] : 0.f;
      H[row][ct * 16 + ccol] = (_Float16)v;
    }
  }
  __syncthreads();
  // GEMM2: (16x256) @ (256x64)
  const _Float16* hrow = &H[lane & 15][0];
  bool full = (m_base + 16 <= MFF);
#pragma unroll
  for (int ct = 0; ct < 4; ++ct) {
    float bv = b2[ct * 16 + ccol];
    v8f acc;
#pragma unroll
    for (int r = 0; r < 8; ++r) acc[r] = bv;
#pragma unroll
    for (int ks = 0; ks < 8; ++ks) {
      v16h a = load_a_f16row(hrow, ks * 32, lane);
      acc = wmma_f16(a, load_b_pack(W2s, 8, ct, ks, lane), acc);
    }
#pragma unroll
    for (int r = 0; r < 8; ++r) {
      int mo = m_base + r + ((lane >> 4) << 3);
      if (full || mo < MFF) {
        int b2i = mo / (Nn * Tn);
        int rem2 = mo % (Nn * Tn);
        int n2 = rem2 / Tn, t2 = rem2 % Tn;
        size_t o = ((size_t)(b2i * Tn + t2) * NP + n2) * Cn + ct * 16 + ccol;
        y3pre[o] = acc[r] + y2[o];
      }
    }
  }
}

extern "C" void kernel_launch(void* const* d_in, const int* in_sizes, int n_in,
                              void* d_out, int out_size, void* d_ws,
                              size_t ws_size, hipStream_t stream) {
  const float* x    = (const float*)d_in[0];
  const float* adj  = (const float*)d_in[1];
  const float* iw   = (const float*)d_in[2];
  const float* Ww   = (const float*)d_in[4];
  const float* Wb   = (const float*)d_in[5];
  const float* aa   = (const float*)d_in[6];
  const float* g1   = (const float*)d_in[7];
  const float* be1  = (const float*)d_in[8];
  const float* Wq   = (const float*)d_in[9];
  const float* Wqb  = (const float*)d_in[10];
  const float* Wk   = (const float*)d_in[11];
  const float* Wkb  = (const float*)d_in[12];
  const float* Wv   = (const float*)d_in[13];
  const float* Wvb  = (const float*)d_in[14];
  const float* Wo   = (const float*)d_in[15];
  const float* Wob  = (const float*)d_in[16];
  const float* g2   = (const float*)d_in[17];
  const float* be2  = (const float*)d_in[18];
  const float* fw1  = (const float*)d_in[19];
  const float* fb1  = (const float*)d_in[20];
  const float* fw2  = (const float*)d_in[21];
  const float* fb2  = (const float*)d_in[22];
  const float* g3   = (const float*)d_in[23];
  const float* be3  = (const float*)d_in[24];

  float* ws    = (float*)d_ws;
  float* Wh    = ws;              // padded; reused as y2pre
  float* y1pre = ws + PSZ;        // padded; reused as ctx
  float* y1    = ws + 2 * PSZ;    // padded; reused as y3pre
  float* y2    = ws + 3 * PSZ;    // padded
  float* Qb    = ws + 4 * PSZ;
  float* Kb    = Qb + SZQ;
  float* Vb    = Kb + SZQ;
  float* fbase = Vb + SZQ;
  _Float16* xt = (_Float16*)fbase;        // BT*NP*64 halves = 983040 floats
  float* wbase = fbase + 983040;
  _Float16* pWw = (_Float16*)wbase;       // 4096 halves each for 64x64 weights
  _Float16* pWq = pWw + 4096;
  _Float16* pWk = pWq + 4096;
  _Float16* pWv = pWk + 4096;
  _Float16* pWo = pWv + 4096;
  _Float16* pW1 = pWo + 4096;             // 16384 halves
  _Float16* pW2 = pW1 + 16384;            // 16384 halves
  float* Sb    = wbase + 32768;           // 6144 used, pad 8192
  float* s1b   = Sb + 8192;
  float* s2b   = s1b + 131072;
  float* stats = s2b + 131072;            // 48 used, pad 64
  float* part  = stats + 64;              // 1024
  float* nbrw  = part + 1024;             // pad 1024
  int*   nbri  = (int*)(nbrw + 1024);

  const int gemm_blocks = (BT * TILES_BT) / 4;  // 480
  const int flat_gat = (BT * Nn * HSn + 255) / 256;
  const int ln_blocks = (int)(((size_t)Bn * NS + 255) / 256);
  const int ff_blocks = (MFT + 3) / 4;
  const int px_blocks = (int)(((size_t)BT * NP * FINn + 255) / 256);

  k_prep<<<(Nn + 127) / 128, 128, 0, stream>>>(adj, iw, nbri, nbrw);
  k_pack_x<<<px_blocks, 256, 0, stream>>>(x, xt);
  k_packw<<<1, 256, 0, stream>>>(Ww, FINn, 4, 2, pWw);
  k_packw<<<1, 256, 0, stream>>>(Wq, Cn, 4, 2, pWq);
  k_packw<<<1, 256, 0, stream>>>(Wk, Cn, 4, 2, pWk);
  k_packw<<<1, 256, 0, stream>>>(Wv, Cn, 4, 2, pWv);
  k_packw<<<1, 256, 0, stream>>>(Wo, Cn, 4, 2, pWo);
  k_packw<<<4, 256, 0, stream>>>(fw1, Cn, 16, 2, pW1);
  k_packw<<<4, 256, 0, stream>>>(fw2, FFn, 4, 8, pW2);

  k_gemm_wh<<<gemm_blocks, 128, 0, stream>>>(xt, pWw, Wb, Wh);
  k_redS<<<BT, 64, 0, stream>>>(Wh, Sb);
  k_scores<<<flat_gat, 256, 0, stream>>>(Wh, aa, s1b, s2b);
  k_gat<<<flat_gat, 256, 0, stream>>>(Wh, Sb, s1b, s2b, nbri, nbrw, x, y1pre);

  k_stats<<<dim3(64, Bn), 256, 0, stream>>>(y1pre, part);
  k_stats_fin<<<Bn, 64, 0, stream>>>(part, stats + 0);
  k_apply_ln<<<ln_blocks, 256, 0, stream>>>(y1pre, g1, be1, stats + 0, y1, 0);

  k_qkv<<<dim3(gemm_blocks, 3), 128, 0, stream>>>(y1, pWq, pWk, pWv, Wqb, Wkb,
                                                  Wvb, Qb, Kb, Vb);
  k_attn<<<flat_gat, 256, 0, stream>>>(Qb, Kb, Vb, /*ctx=*/y1pre);
  k_gemm_o<<<gemm_blocks, 128, 0, stream>>>(/*ctx=*/y1pre, pWo, Wob, y1,
                                            /*y2pre=*/Wh);

  k_stats<<<dim3(64, Bn), 256, 0, stream>>>(Wh, part);
  k_stats_fin<<<Bn, 64, 0, stream>>>(part, stats + 16);
  k_apply_ln<<<ln_blocks, 256, 0, stream>>>(Wh, g2, be2, stats + 16, y2, 0);

  k_ff<<<ff_blocks, 128, 0, stream>>>(y2, pW1, fb1, pW2, fb2, /*y3pre=*/y1);

  k_stats<<<dim3(64, Bn), 256, 0, stream>>>(y1, part);
  k_stats_fin<<<Bn, 64, 0, stream>>>(part, stats + 32);
  k_apply_ln<<<ln_blocks, 256, 0, stream>>>(y1, g3, be3, stats + 32,
                                            (float*)d_out, 1);
}
